// Attention_40767829574369
// MI455X (gfx1250) — compile-verified, matrix operand order checked by reading, not run on hardware
//
// MI455X / gfx1250 implementation of the TTS alignment pipeline.
//
// Heavy math (conv2: 19 GFLOP, conv3: 75 GFLOP, 3 GEMMs: ~5 GFLOP) runs on
// v_wmma_f32_16x16x32_f16 (f16 multiplicands, f32 accumulation). Fragment
// data is laid out so each lane's 16 halves are two contiguous 16B runs
// (-> ds_load_b128 / global_load_b128), instead of 16 scalar gathers.
// Everything else (conv1 with Cin=1, sequential monotonic-path DP, masked
// softmax losses) is latency/bandwidth trivial (<0.5 GB traffic vs 23.3 TB/s
// HBM) and runs on VALU with wave32 reductions. All reductions are
// fixed-order (deterministic); no atomics; all scratch in d_ws.

#include <hip/hip_runtime.h>
#include <hip/hip_bf16.h>

typedef __attribute__((ext_vector_type(16))) _Float16 v16h;
typedef __attribute__((ext_vector_type(8)))  _Float16 v8h;
typedef __attribute__((ext_vector_type(8)))  float    v8f;

#define WMMA_F16(a, b, c) \
  __builtin_amdgcn_wmma_f32_16x16x32_f16(false, (a), false, (b), (short)0, (c), false, false)

#define NB   32
#define TT   128    // text length
#define TD   1600   // spec time steps
#define MEL  80
#define EH   512
#define AH   128
#define TQ   200    // query time steps (800/4)
#define W1   800
#define H1   40
#define C1   32
#define H2   20
#define C2   64
#define C3   128
#define NEGF (-1.0e9f)
#define BN_SCALE 0.9999950000374997f   // 1/sqrt(1+1e-5)

// f16 WMMA A/B fragment: element e (0..15) of lane L holds K index:
//   k = (e>>3)*16 + (L>>4)*8 + (e&7)          (ISA 7.12.2, 16-bit A 16x32)
// A: lane L -> row M = L&15.  B (symmetric): lane L -> col N = L&15.
__device__ __forceinline__ int frag_k(int lane, int e) {
  return ((e >> 3) << 4) + ((lane >> 4) << 3) + (e & 7);
}

// Fragment from 32 k-contiguous halves at `base` (16B-aligned): two 8-half
// runs at +hi8 and +16+hi8 -> two ds_load_b128, no scalar gathers.
__device__ __forceinline__ v16h load_frag(const _Float16* base, int hi8) {
  v8h lo = *(const v8h*)(base + hi8);
  v8h hi = *(const v8h*)(base + 16 + hi8);
  return __builtin_shufflevector(lo, hi, 0, 1, 2, 3, 4, 5, 6, 7,
                                         8, 9, 10, 11, 12, 13, 14, 15);
}

__device__ __forceinline__ v16h cvt16(float4 x0, float4 x1, float4 x2, float4 x3) {
  v16h r;
  r[0]  = (_Float16)x0.x; r[1]  = (_Float16)x0.y; r[2]  = (_Float16)x0.z; r[3]  = (_Float16)x0.w;
  r[4]  = (_Float16)x1.x; r[5]  = (_Float16)x1.y; r[6]  = (_Float16)x1.z; r[7]  = (_Float16)x1.w;
  r[8]  = (_Float16)x2.x; r[9]  = (_Float16)x2.y; r[10] = (_Float16)x2.z; r[11] = (_Float16)x2.w;
  r[12] = (_Float16)x3.x; r[13] = (_Float16)x3.y; r[14] = (_Float16)x3.z; r[15] = (_Float16)x3.w;
  return r;
}

// ---------------------------------------------------------------------------
// conv1: Cin=1, 3x3, stride (2,2), pad (t=0,l=0). K-dim is only 9 -> VALU.
// input x[n][0][mel][t] = spec[n][t][mel], masked by t < spec_len.
// ---------------------------------------------------------------------------
__global__ void conv1_kernel(const float* __restrict__ spec,
                             const float* __restrict__ w,
                             const float* __restrict__ b,
                             const float* __restrict__ g,
                             const float* __restrict__ be,
                             const int* __restrict__ spec_len,
                             float* __restrict__ out) {
  long idx = (long)blockIdx.x * blockDim.x + threadIdx.x;
  const long total = (long)NB * C1 * H1 * W1;
  if (idx >= total) return;
  int wq = idx % W1;
  int h  = (int)((idx / W1) % H1);
  int co = (int)((idx / ((long)W1 * H1)) % C1);
  int n  = (int)(idx / ((long)W1 * H1 * C1));
  int L = spec_len[n];
  float acc = 0.f;
  for (int kh = 0; kh < 3; ++kh) {
    int hi = 2 * h + kh;
    if (hi >= MEL) continue;
    for (int kw = 0; kw < 3; ++kw) {
      int wi = 2 * wq + kw;
      if (wi >= TD || wi >= L) continue;
      acc += spec[((long)n * TD + wi) * MEL + hi] * w[co * 9 + kh * 3 + kw];
    }
  }
  float v = (acc + b[co]) * (g[co] * BN_SCALE) + be[co];
  out[idx] = v > 0.f ? v : 0.f;
}

// ---------------------------------------------------------------------------
// conv2/conv3 as implicit GEMM on WMMA.
// Block = 5 waves (160 thr); computes 16 out-channels x 80 out-columns at a
// fixed output row h. Weights staged fragment-major (lane-contiguous 32B),
// input strip staged channel-contiguous per column. Per wave: CIN/32 chunks
// x 9 taps WMMAs accumulate a 16x16 f32 tile.
// ---------------------------------------------------------------------------
template <int CIN, int COUT, int HIN, int HOUT, int SH, int PT>
__global__ __launch_bounds__(160)
void conv_wmma_kernel(const float* __restrict__ in,
                      const float* __restrict__ wgt,   // (COUT,CIN,3,3)
                      const float* __restrict__ bias,
                      const float* __restrict__ gma,
                      const float* __restrict__ beta,
                      const int* __restrict__ spec_len,
                      float* __restrict__ out) {
  constexpr int STRIP = 80;
  constexpr int COLS  = STRIP + 2;          // left/right halo (pad_l = 1)
  constexpr int NCH   = CIN / 32;           // K chunks of 32
  __shared__ alignas(32) _Float16 Wf[NCH * 9 * 32 * 16];  // fragment-major A
  __shared__ alignas(32) _Float16 Xl[3 * COLS * CIN];     // [kh][col][ci]

  int n   = blockIdx.z;
  int co0 = blockIdx.y * 16;
  int h   = blockIdx.x / 10;
  int wb  = (blockIdx.x % 10) * STRIP;
  int L   = (spec_len[n] + 1) >> 1;         // mask width after stage-1 stride 2

  // A fragments pre-formed: Wf[((chunk*9+tap)*32+lane)*16+e]
  for (int i = threadIdx.x; i < NCH * 9 * 512; i += 160) {
    int e     = i & 15;
    int ln    = (i >> 4) & 31;
    int tap   = (i >> 9) % 9;
    int chunk = i / (9 * 512);
    int m  = ln & 15;
    int ci = chunk * 32 + frag_k(ln, e);
    Wf[i] = (_Float16)wgt[((long)(co0 + m) * CIN + ci) * 9 + tap];
  }
  // Input strip: global reads column-fastest (coalesced), LDS channel-contig.
  for (int i = threadIdx.x; i < CIN * 3 * COLS; i += 160) {
    int col = i % COLS;
    int r   = (i / COLS) % 3;
    int ci  = i / (3 * COLS);
    int hin = h * SH + r - PT;
    int win = wb - 1 + col;
    float v = 0.f;
    if (hin >= 0 && hin < HIN && win >= 0 && win < W1 && win < L)
      v = in[(((long)n * CIN + ci) * HIN + hin) * W1 + win];
    Xl[(r * COLS + col) * CIN + ci] = (_Float16)v;
  }
  __syncthreads();

  int wave = threadIdx.x >> 5;
  int lane = threadIdx.x & 31;
  int ml   = lane & 15;                      // A row / B col / D col
  int hi8  = (lane >> 4) << 3;
  v8f acc = {};
  for (int chunk = 0; chunk < NCH; ++chunk) {
    for (int kh = 0; kh < 3; ++kh) {
      for (int kw = 0; kw < 3; ++kw) {
        v16h a = *(const v16h*)&Wf[((chunk * 9 + kh * 3 + kw) * 32 + lane) * 16];
        int col = wave * 16 + ml + kw;
        v16h b = load_frag(&Xl[(kh * COLS + col) * CIN + chunk * 32], hi8);
        acc = WMMA_F16(a, b, acc);
      }
    }
  }
  int hi16 = lane >> 4;
#pragma unroll
  for (int r = 0; r < 8; ++r) {
    int co = co0 + r + 8 * hi16;
    int wq = wb + wave * 16 + ml;
    float v = acc[r] + bias[co];
    v = v * (gma[co] * BN_SCALE) + beta[co];
    v = v > 0.f ? v : 0.f;
    out[(((long)n * COUT + co) * HOUT + h) * W1 + wq] = v;
  }
}

// ---------------------------------------------------------------------------
// Generic WMMA GEMM: C(M x 128) = A(M x K) * B(K x 128) + bias.
// Block = 8 waves (256 thr): 16 M-rows x full 128 N-cols; K chunked by 32.
// A chunk staged fragment-major; B chunk staged transposed (Bt[col][k]) so
// lane fragments are contiguous b128 LDS loads. MODE is compile-time so the
// staging address math is straight-line (no exec-mask branch ladder).
// MODE 0: A row-major (lda).  MODE 1: query im2col view of conv3 output:
//   A[m][k] = conv3[(m/200)*2560*800 + k*800 + (m%200)*4]
// ---------------------------------------------------------------------------
template <int MODE>
__global__ __launch_bounds__(256)
void gemm_wmma_kernel(const float* __restrict__ A,
                      const float* __restrict__ B,
                      const float* __restrict__ bias,
                      float* __restrict__ C,
                      int K, long lda) {
  __shared__ alignas(32) _Float16 Af[512];        // fragment-major A chunk
  __shared__ alignas(32) _Float16 Bt[128 * 32];   // [col][k]
  int m0 = blockIdx.x * 16;
  int wave = threadIdx.x >> 5, lane = threadIdx.x & 31;
  int ml = lane & 15;
  int hi8 = (lane >> 4) << 3;
  v8f acc = {};
  for (int k0 = 0; k0 < K; k0 += 32) {
    __syncthreads();
#pragma unroll
    for (int ii = 0; ii < 2; ++ii) {
      int i  = threadIdx.x + ii * 256;
      int e  = i & 15;
      int ln = (i >> 4) & 31;
      int gm = m0 + (ln & 15), gk = k0 + frag_k(ln, e);
      long addr = (MODE == 0)
          ? (long)gm * lda + gk
          : (long)(gm / 200) * (2560L * 800) + (long)gk * 800 + (gm % 200) * 4;
      Af[i] = (_Float16)A[addr];
    }
#pragma unroll
    for (int ii = 0; ii < 16; ++ii) {
      int i = threadIdx.x + ii * 256;
      int col = i & 127, k = i >> 7;         // global reads coalesced over col
      Bt[col * 32 + k] = (_Float16)B[(long)(k0 + k) * 128 + col];
    }
    __syncthreads();
    v16h a = *(const v16h*)&Af[lane * 16];
    v16h b = load_frag(&Bt[(wave * 16 + ml) * 32], hi8);
    acc = WMMA_F16(a, b, acc);
  }
  int hi16 = lane >> 4;
#pragma unroll
  for (int r = 0; r < 8; ++r) {
    int m = m0 + r + 8 * hi16;
    int col = wave * 16 + ml;
    C[(long)m * 128 + col] = acc[r] + bias[col];
  }
}

// Row L2-normalization of (rows x 128) matrix; one wave per row.
__global__ __launch_bounds__(256)
void rownorm_kernel(const float* __restrict__ X, float* __restrict__ Y, int rows) {
  int row = blockIdx.x * 8 + (threadIdx.x >> 5);
  if (row >= rows) return;
  int lane = threadIdx.x & 31;
  float v[4], s = 0.f;
  for (int i = 0; i < 4; ++i) { v[i] = X[(long)row * 128 + lane + 32 * i]; s += v[i] * v[i]; }
  for (int off = 16; off > 0; off >>= 1) s += __shfl_xor(s, off, 32);
  float inv = 1.f / sqrtf(s);
  for (int i = 0; i < 4; ++i) Y[(long)row * 128 + lane + 32 * i] = v[i] * inv;
}

// ---------------------------------------------------------------------------
// Batched similarity GEMM: logit[n][t][q] = (key_n[n,t,:] . q_n[n,q,:])*w+b.
// One wave per 16x16 (t,q) tile. Fragments loaded as float4 (b128) straight
// from global (rows are k-contiguous); out-of-range q clamped: the garbage
// D-columns are never stored, and each D column depends only on its own B
// column, so no zero-masking is required (EXEC stays all-ones).
// ---------------------------------------------------------------------------
__global__ __launch_bounds__(32)
void logit_wmma_kernel(const float* __restrict__ Kn, const float* __restrict__ Qn,
                       const float* __restrict__ wsim, const float* __restrict__ bsim,
                       float* __restrict__ Lg) {
  int n = blockIdx.z;
  int t0 = blockIdx.x * 16;
  int q0 = blockIdx.y * 16;
  int lane = threadIdx.x & 31;
  int ml = lane & 15;
  int hi8 = (lane >> 4) << 3;
  int qe = q0 + ml; if (qe > TQ - 1) qe = TQ - 1;
  const float* ap = Kn + ((long)n * TT + (t0 + ml)) * AH;
  const float* bp = Qn + ((long)n * TQ + qe) * AH;
  v8f acc = {};
#pragma unroll
  for (int k0 = 0; k0 < AH; k0 += 32) {
    const float* a0 = ap + k0 + hi8;
    const float* b0 = bp + k0 + hi8;
    v16h a = cvt16(*(const float4*)(a0),      *(const float4*)(a0 + 4),
                   *(const float4*)(a0 + 16), *(const float4*)(a0 + 20));
    v16h b = cvt16(*(const float4*)(b0),      *(const float4*)(b0 + 4),
                   *(const float4*)(b0 + 16), *(const float4*)(b0 + 20));
    acc = WMMA_F16(a, b, acc);
  }
  float ws = wsim[0], bs = bsim[0];
  int hi16 = lane >> 4;
#pragma unroll
  for (int r = 0; r < 8; ++r) {
    int t = t0 + r + 8 * hi16;
    int q = q0 + ml;
    if (q < TQ) Lg[((long)n * TT + t) * TQ + q] = acc[r] * ws + bs;
  }
}

// ---------------------------------------------------------------------------
// Monotonic alignment DP (maximum_path). One block per batch, thread = state x.
// Forward stores pre-update Q rows (Qm1) to ws; thread 0 runs the traceback.
// ---------------------------------------------------------------------------
__global__ __launch_bounds__(128)
void dp_kernel(const float* __restrict__ Lg, const int* __restrict__ tlen,
               const int* __restrict__ slen, float* __restrict__ Qm1,
               int* __restrict__ sel) {
  __shared__ float qs[TT + 1];
  __shared__ int ssel[TQ];
  int n = blockIdx.x;
  int x = threadIdx.x;
  int tx = tlen[n];
  int sl = slen[n] >> 3;                    // spec_lengths // 8
  const float* Lb = Lg + (long)n * TT * TQ + (long)x * TQ;
  float q = (x == 0) ? 0.f : NEGF;
  for (int y = 0; y < TQ; ++y) {
    Qm1[((long)n * TQ + y) * TT + x] = q;   // Q[y-1] (q0 at y==0)
    qs[x + 1] = q;
    if (x == 0) qs[0] = NEGF;
    __syncthreads();
    float shifted = qs[x];
    float v = (x < tx && y < sl) ? expf(Lb[y]) : 0.f;
    float nq = v + fmaxf(q, shifted);
    if (x > y) nq = NEGF;
    q = nq;
    __syncthreads();
  }
  if (x == 0) {
    int idx = tx - 1;
    for (int y = TQ - 1; y >= 0; --y) {
      bool active = y < sl;
      ssel[y] = active ? idx : -1;
      const float* qm1 = Qm1 + ((long)n * TQ + y) * TT;
      float vs = qm1[idx];
      float vm = qm1[idx > 0 ? idx - 1 : 0];
      bool move = (idx != 0) && ((idx == y) || (vs < vm));
      if (active && move) --idx;
    }
  }
  __syncthreads();
  for (int y = x; y < TQ; y += 128) sel[n * TQ + y] = ssel[y];
}

// Per-(n,q) loss-1 partials + sharpness partials. One wave per (n,q).
__global__ __launch_bounds__(256)
void loss1_kernel(const float* __restrict__ Lg, const int* __restrict__ tlen,
                  const int* __restrict__ slen, const int* __restrict__ sel,
                  float* __restrict__ l1p, float* __restrict__ sharpp) {
  int gw = blockIdx.x * 8 + (threadIdx.x >> 5);
  if (gw >= NB * TQ) return;
  int lane = threadIdx.x & 31;
  int n = gw / TQ, q = gw % TQ;
  int tx = tlen[n], sl = slen[n] >> 3;
  const float* Lb = Lg + (long)n * TT * TQ + q;
  float lv[4], m1 = -1e30f;
  for (int i = 0; i < 4; ++i) { lv[i] = Lb[(long)(lane + 32 * i) * TQ]; m1 = fmaxf(m1, lv[i]); }
  for (int off = 16; off > 0; off >>= 1) m1 = fmaxf(m1, __shfl_xor(m1, off, 32));
  float S1 = 0.f, mm = -1e30f;
  for (int i = 0; i < 4; ++i) {
    int t = lane + 32 * i;
    if (t < tx) { float e = expf(lv[i] - m1); S1 += e; mm = fmaxf(mm, e); }
  }
  for (int off = 16; off > 0; off >>= 1) {
    S1 += __shfl_xor(S1, off, 32);
    mm = fmaxf(mm, __shfl_xor(mm, off, 32));
  }
  if (lane == 0) {
    bool active = q < sl;
    int s = sel[n * TQ + q];
    float net_sel = active ? (Lb[(long)s * TQ] - m1) : 0.f;
    l1p[gw]    = active ? (-net_sel + logf(fmaxf(S1, 1e-30f))) : 0.f;
    float mc   = active ? (mm / fmaxf(S1, 1e-12f)) : 0.f;
    sharpp[gw] = 1.f - mc;
  }
}

// Per-(n,t) loss-2 partials. One wave per (n,t).
__global__ __launch_bounds__(256)
void loss2_kernel(const float* __restrict__ Lg, const int* __restrict__ tlen,
                  const int* __restrict__ slen, const int* __restrict__ sel,
                  float* __restrict__ l2p) {
  int gw = blockIdx.x * 8 + (threadIdx.x >> 5);
  if (gw >= NB * TT) return;
  int lane = threadIdx.x & 31;
  int n = gw / TT, t = gw % TT;
  int tx = tlen[n], sl = slen[n] >> 3;
  const float* Lb = Lg + ((long)n * TT + t) * TQ;
  const int* sb = sel + n * TQ;
  float m2 = -1e30f;
  for (int q = lane; q < TQ; q += 32) m2 = fmaxf(m2, Lb[q]);
  for (int off = 16; off > 0; off >>= 1) m2 = fmaxf(m2, __shfl_xor(m2, off, 32));
  float A2 = 0.f, B2 = 0.f;
  for (int q = lane; q < TQ; q += 32) {
    float e = expf(Lb[q] - m2);
    if (q < sl) B2 += e;
    if (sb[q] == t) A2 += e;
  }
  for (int off = 16; off > 0; off >>= 1) {
    A2 += __shfl_xor(A2, off, 32);
    B2 += __shfl_xor(B2, off, 32);
  }
  if (lane == 0)
    l2p[gw] = (t < tx) ? (-logf(fmaxf(A2, 1e-30f)) + logf(fmaxf(B2, 1e-30f))) : 0.f;
}

// attention_out (repeat x2 along q) and att_mask_out (int32).
__global__ void writeout_kernel(const int* __restrict__ sel, const int* __restrict__ tlen,
                                const int* __restrict__ slen, float* __restrict__ att_out,
                                int* __restrict__ mask_out) {
  long idx = (long)blockIdx.x * blockDim.x + threadIdx.x;
  if (idx >= (long)NB * TT * 400) return;
  int yy = (int)(idx % 400);
  int t  = (int)((idx / 400) % TT);
  int n  = (int)(idx / (400L * TT));
  int y  = yy >> 1;
  att_out[idx]  = (sel[n * TQ + y] == t) ? 1.f : 0.f;
  mask_out[idx] = (t < tlen[n] && y < (slen[n] >> 3)) ? 1 : 0;
}

// Deterministic final reduction (single block) -> scalars.
__global__ __launch_bounds__(256)
void finalize_kernel(const float* __restrict__ l1p, const float* __restrict__ sharpp,
                     const float* __restrict__ l2p, const int* __restrict__ tlen,
                     const int* __restrict__ slen, float* __restrict__ out_loss,
                     float* __restrict__ out_sharp, float* __restrict__ out_nll) {
  __shared__ float red[256];
  int tid = threadIdx.x;
  float s1 = 0.f, s2 = 0.f, s3 = 0.f;
  for (int i = tid; i < NB * TQ; i += 256) { s1 += l1p[i]; s3 += sharpp[i]; }
  for (int i = tid; i < NB * TT; i += 256) s2 += l2p[i];
  red[tid] = s1; __syncthreads();
  for (int o = 128; o > 0; o >>= 1) { if (tid < o) red[tid] += red[tid + o]; __syncthreads(); }
  float suml1 = red[0]; __syncthreads();
  red[tid] = s2; __syncthreads();
  for (int o = 128; o > 0; o >>= 1) { if (tid < o) red[tid] += red[tid + o]; __syncthreads(); }
  float suml2 = red[0]; __syncthreads();
  red[tid] = s3; __syncthreads();
  for (int o = 128; o > 0; o >>= 1) { if (tid < o) red[tid] += red[tid + o]; __syncthreads(); }
  float sumsh = red[0];
  if (tid == 0) {
    int ssl = 0, stx = 0;
    for (int n = 0; n < NB; ++n) { ssl += slen[n] >> 3; stx += tlen[n]; }
    float nll   = suml1 / (float)ssl;
    float loss2 = suml2 / (float)stx;
    out_loss[0]  = nll + 0.1f * loss2;
    out_sharp[0] = sumsh / (float)(NB * TQ);
    out_nll[0]   = nll;
  }
}

// ---------------------------------------------------------------------------
extern "C" void kernel_launch(void* const* d_in, const int* in_sizes, int n_in,
                              void* d_out, int out_size, void* d_ws, size_t ws_size,
                              hipStream_t stream) {
  const float* text   = (const float*)d_in[0];
  const float* spec   = (const float*)d_in[1];
  const int*   tlen   = (const int*)d_in[2];
  const int*   slen   = (const int*)d_in[3];
  const float* w_enc  = (const float*)d_in[4];
  const float* b_enc  = (const float*)d_in[5];
  const float* cw1    = (const float*)d_in[6];
  const float* cb1    = (const float*)d_in[7];
  const float* g1     = (const float*)d_in[8];
  const float* be1    = (const float*)d_in[9];
  const float* cw2    = (const float*)d_in[10];
  const float* cb2    = (const float*)d_in[11];
  const float* g2     = (const float*)d_in[12];
  const float* be2    = (const float*)d_in[13];
  const float* cw3    = (const float*)d_in[14];
  const float* cb3    = (const float*)d_in[15];
  const float* g3     = (const float*)d_in[16];
  const float* be3    = (const float*)d_in[17];
  const float* w_q    = (const float*)d_in[18];
  const float* b_q    = (const float*)d_in[19];
  const float* w_sim  = (const float*)d_in[20];
  const float* b_sim  = (const float*)d_in[21];

  char* ws = (char*)d_ws;
  // Large conv regions (convA reused: conv1-out then conv3-out).
  float* convA = (float*)(ws);                         // 262,144,000 B
  float* convB = (float*)(ws + 262144000L);            // 131,072,000 B
  char*  sm    = ws + 393216000L;                      // small region
  float* key    = (float*)(sm + 0L);                   // 32*128*128
  float* key_n  = (float*)(sm + 2097152L);
  float* query  = (float*)(sm + 4194304L);             // 32*200*128
  float* q_n    = (float*)(sm + 7471104L);
  float* logit  = (float*)(sm + 10747904L);            // 32*128*200
  float* Qm1    = (float*)(sm + 17301504L);            // 32*200*128
  int*   sel    = (int*)  (sm + 20578304L);            // 32*200
  float* l1p    = (float*)(sm + 20603904L);            // 6400
  float* sharpp = (float*)(sm + 20629504L);            // 6400
  float* l2p    = (float*)(sm + 20655104L);            // 4096

  float* outf    = (float*)d_out;
  float* att_out = outf;                               // 1,638,400
  float* o_loss  = outf + 1638400;
  int*   mask_o  = (int*)d_out + 1638401;              // 1,638,400
  float* o_sharp = outf + 3276801;
  float* o_nll   = outf + 3276802;

  // 1) key = text @ w_enc + b_enc   (M=4096, K=512)
  gemm_wmma_kernel<0><<<256, 256, 0, stream>>>(text, w_enc, b_enc, key, 512, 512);
  rownorm_kernel<<<512, 256, 0, stream>>>(key, key_n, 4096);

  // 2) conv stack
  conv1_kernel<<<128000, 256, 0, stream>>>(spec, cw1, cb1, g1, be1, slen, convA);
  conv_wmma_kernel<C1, C2, H1, H2, 2, 0>
      <<<dim3(200, 4, 32), 160, 0, stream>>>(convA, cw2, cb2, g2, be2, slen, convB);
  conv_wmma_kernel<C2, C3, H2, H2, 1, 1>
      <<<dim3(200, 8, 32), 160, 0, stream>>>(convB, cw3, cb3, g3, be3, slen, convA);

  // 3) query = im2col(conv3)[0::4] @ w_q + b_q   (M=6400, K=2560)
  gemm_wmma_kernel<1><<<400, 256, 0, stream>>>(convA, w_q, b_q, query, 2560, 0);
  rownorm_kernel<<<800, 256, 0, stream>>>(query, q_n, 6400);

  // 4) logit[n,t,q] = (key_n . q_n) * w_sim + b_sim
  logit_wmma_kernel<<<dim3(8, 13, 32), 32, 0, stream>>>(key_n, q_n, w_sim, b_sim, logit);

  // 5) monotonic alignment DP + traceback
  dp_kernel<<<32, 128, 0, stream>>>(logit, tlen, slen, Qm1, sel);

  // 6) losses + outputs
  loss1_kernel<<<800, 256, 0, stream>>>(logit, tlen, slen, sel, l1p, sharpp);
  loss2_kernel<<<512, 256, 0, stream>>>(logit, tlen, slen, sel, l2p);
  writeout_kernel<<<6400, 256, 0, stream>>>(sel, tlen, slen, att_out, mask_o);
  finalize_kernel<<<1, 256, 0, stream>>>(l1p, sharpp, l2p, tlen, slen, o_loss, o_sharp, o_nll);

  (void)in_sizes; (void)n_in; (void)out_size; (void)ws_size;
}
